// SP_Encoder_69664369541844
// MI455X (gfx1250) — matrix operand order repaired and checked by reading, not compiled, not run on hardware
//
#include <hip/hip_runtime.h>
#include <stdint.h>

typedef __attribute__((ext_vector_type(16))) __bf16 v16bf;
typedef __attribute__((ext_vector_type(8)))  float  v8f;

#define HH   64
#define DIN  768
#define BB   64
#define SS   512
#define NG   512   // 2 dirs * 4H gate columns

// ---------- helpers ----------
__device__ __forceinline__ unsigned short f2bf(float f) {
    union { float f; unsigned u; } x; x.f = f;
    unsigned r = x.u + 0x7FFFu + ((x.u >> 16) & 1u);   // round-to-nearest-even
    return (unsigned short)(r >> 16);
}
__device__ __forceinline__ float sigf(float x) { return 1.0f / (1.0f + __expf(-x)); }

union ABu { uint4 q[2]; v16bf v; };

// A fragment: 16x32 bf16 tile, row-major src (ld elems). ISA layout:
// lane L: row = L&15; elems 0..7 -> K = k0+8g+0..7 ; elems 8..15 -> K = k0+16+8g+0..7
__device__ __forceinline__ v16bf load_afrag(const unsigned short* base, int ld, int k0, int lane) {
    int r = lane & 15, g = lane >> 4;
    const unsigned short* p = base + (size_t)r * ld + k0 + 8 * g;
    ABu u;
    u.q[0] = *(const uint4*)(p);
    u.q[1] = *(const uint4*)(p + 16);
    return u.v;
}
// B fragment: 32x16 bf16 tile from row-major weight W[n][k] (ld elems).
// lane L: col n = n0+(L&15); elems 0..7 -> K = k0+8g+0..7 ; elems 8..15 -> K = k0+16+8g+0..7
__device__ __forceinline__ v16bf load_bfrag(const unsigned short* w, int ld, int n0, int k0, int lane) {
    int n = n0 + (lane & 15), g = lane >> 4;
    const unsigned short* p = w + (size_t)n * ld + k0 + 8 * g;
    ABu u;
    u.q[0] = *(const uint4*)(p);
    u.q[1] = *(const uint4*)(p + 16);
    return u.v;
}

// ---------- utility kernels ----------
__global__ void fill_u16(unsigned short* p, unsigned short v, size_t n) {
    size_t i = (size_t)blockIdx.x * blockDim.x + threadIdx.x;
    if (i < n) p[i] = v;
}
__global__ void fill_f32(float* p, float v, size_t n) {
    size_t i = (size_t)blockIdx.x * blockDim.x + threadIdx.x;
    if (i < n) p[i] = v;
}
__global__ void cvt_bf16(const float* __restrict__ in, unsigned short* __restrict__ out, size_t n) {
    size_t i = (size_t)blockIdx.x * blockDim.x + threadIdx.x;
    if (i < n) out[i] = f2bf(in[i]);
}

// ---------- subword -> word segment-sum pooling (+mask) ----------
// word_ids are sorted per batch row, so a running accumulator with flush-on-change
// implements segment_sum without atomics. One block per batch, 768 feature lanes.
__global__ __launch_bounds__(768) void pool_kernel(
    const float* __restrict__ emb,          // [B,S,768]
    const int*   __restrict__ wid,          // [B,S]
    unsigned short* __restrict__ merged,    // [B,S,768] bf16, pre-zeroed
    float* __restrict__ mask)               // [B,S], pre-zeroed
{
    int b = blockIdx.x;
    int dI = threadIdx.x;
    const int* w = wid + (size_t)b * SS;
    const float* e = emb + (size_t)b * SS * DIN + dI;
    unsigned short* m = merged + (size_t)b * SS * DIN + dI;
    float acc = 0.0f;
    int cur = w[0];
    for (int s = 0; s < SS; ++s) {
        int id = w[s];
        if (id != cur) {
            m[(size_t)cur * DIN] = f2bf(acc);
            if (dI == 0) mask[(size_t)b * SS + cur] = 1.0f;
            acc = 0.0f;
            cur = id;
        }
        acc += e[(size_t)s * DIN];
    }
    m[(size_t)cur * DIN] = f2bf(acc);
    if (dI == 0) mask[(size_t)b * SS + cur] = 1.0f;
}

// ---------- time-parallel input GEMM: pre = A @ W^T + bias ----------
// A: [M=32768, K] bf16 row-major; W: [512, K] bf16 row-major; C: [M, 512] f32.
// 8 waves/block: each wave -> 16 rows x 64 cols (4 N-tiles), K-loop in steps of 32.
__global__ __launch_bounds__(256) void gemm_pre(
    const unsigned short* __restrict__ A, int lda,
    const unsigned short* __restrict__ W,
    const float* __restrict__ bias,
    float* __restrict__ C,
    int K)
{
    int lane = threadIdx.x & 31;
    int wave = threadIdx.x >> 5;                 // 0..7
    int m0 = blockIdx.x * 128 + wave * 16;
    int n0 = blockIdx.y * 64;
    v8f acc[4] = {};
    const unsigned short* Arow = A + (size_t)m0 * lda;
    for (int k0 = 0; k0 < K; k0 += 32) {
        v16bf a = load_afrag(Arow, lda, k0, lane);
#pragma unroll
        for (int j = 0; j < 4; ++j) {
            v16bf b = load_bfrag(W, K, n0 + j * 16, k0, lane);
            acc[j] = __builtin_amdgcn_wmma_f32_16x16x32_bf16(
                false, a, false, b, (short)0, acc[j], false, false);
        }
    }
    int col = lane & 15, g = lane >> 4;
#pragma unroll
    for (int j = 0; j < 4; ++j) {
        int n = n0 + j * 16 + col;
        float bv = bias[n];
#pragma unroll
        for (int r = 0; r < 8; ++r)
            C[(size_t)(m0 + r + 8 * g) * NG + n] = acc[j][r] + bv;
    }
}

// ---------- recurrent bidirectional LSTM layer ----------
// grid (4 batch-tiles, 2 dirs), 512 threads = 16 waves. W_hh B-fragments stay in
// VGPRs for all 512 steps; h (bf16) and c (f32) live in LDS. Per step: 2 WMMAs per
// wave for the 64x256 gate GEMM, then an elementwise gate phase through LDS.
__global__ __launch_bounds__(512) void lstm_rec(
    const float* __restrict__ pre,              // [B*S, 512]
    const unsigned short* __restrict__ Whh,     // [2, 256, 64] bf16 (this layer)
    float* __restrict__ x,                      // [B*S, 128] residual in / out
    unsigned short* __restrict__ xbf,           // [B*S, 128] bf16 copy for next GEMM
    int residual)
{
    __shared__ __align__(16) unsigned short hbuf[16 * HH];  // h (bf16)
    __shared__ __align__(16) float gbuf[16 * 256];          // raw gates
    __shared__ __align__(16) float cbuf[16 * HH];           // cell state

    int tid  = threadIdx.x;
    int lane = tid & 31;
    int wave = tid >> 5;                        // 0..15 -> N-tile
    int b0 = blockIdx.x * 16;
    int d  = blockIdx.y;
    const unsigned short* Wd = Whh + (size_t)d * 256 * HH;

    // persistent B fragments (K = 64 -> two k-steps)
    v16bf bf0 = load_bfrag(Wd, HH, wave * 16, 0,  lane);
    v16bf bf1 = load_bfrag(Wd, HH, wave * 16, 32, lane);

    for (int i = tid; i < 16 * HH; i += 512) { hbuf[i] = 0; cbuf[i] = 0.0f; }
    __syncthreads();

    for (int ti = 0; ti < SS; ++ti) {
        int t = d ? (SS - 1 - ti) : ti;

        v16bf a0 = load_afrag(hbuf, HH, 0,  lane);
        v16bf a1 = load_afrag(hbuf, HH, 32, lane);
        v8f acc = {};
        acc = __builtin_amdgcn_wmma_f32_16x16x32_bf16(false, a0, false, bf0, (short)0, acc, false, false);
        acc = __builtin_amdgcn_wmma_f32_16x16x32_bf16(false, a1, false, bf1, (short)0, acc, false, false);

        int col = lane & 15, g = lane >> 4;
#pragma unroll
        for (int r = 0; r < 8; ++r)
            gbuf[(r + 8 * g) * 256 + wave * 16 + col] = acc[r];
        __syncthreads();

#pragma unroll
        for (int e0 = 0; e0 < 2; ++e0) {
            int e = tid + e0 * 512;                 // 1024 (row,col) cells
            int row = e >> 6, c = e & 63;
            const float* pb = pre + ((size_t)(b0 + row) * SS + t) * NG + d * 256;
            float gi = gbuf[row * 256 +        c] + pb[c];
            float gf = gbuf[row * 256 +  64 +  c] + pb[64 + c];
            float gg = gbuf[row * 256 + 128 +  c] + pb[128 + c];
            float go = gbuf[row * 256 + 192 +  c] + pb[192 + c];
            float cc = sigf(gf) * cbuf[row * HH + c] + sigf(gi) * tanhf(gg);
            float hh = sigf(go) * tanhf(cc);
            cbuf[row * HH + c] = cc;
            hbuf[row * HH + c] = f2bf(hh);
            size_t oi = ((size_t)(b0 + row) * SS + t) * 128 + d * HH + c;
            float ov = hh + (residual ? x[oi] : 0.0f);
            x[oi]   = ov;
            xbf[oi] = f2bf(ov);
        }
        __syncthreads();
    }
}

// ---------- final concat: [x | sn_word_len] ----------
__global__ __launch_bounds__(128) void finalize_k(
    const float* __restrict__ x, const float* __restrict__ wl, float* __restrict__ out)
{
    size_t bs = blockIdx.x;
    int t = threadIdx.x;
    out[bs * 129 + t] = x[bs * 128 + t];
    if (t == 0) out[bs * 129 + 128] = wl[bs];
}

// ---------- launch ----------
extern "C" void kernel_launch(void* const* d_in, const int* in_sizes, int n_in,
                              void* d_out, int out_size, void* d_ws, size_t ws_size,
                              hipStream_t stream) {
    (void)in_sizes; (void)n_in; (void)out_size; (void)ws_size;
    const float* emb   = (const float*)d_in[0];   // [64,512,768]
    const float* wlen  = (const float*)d_in[1];   // [64,512]
    const float* Wih1  = (const float*)d_in[2];   // [2,256,768]
    const float* Whh1  = (const float*)d_in[3];   // [2,256,64]
    const float* b1    = (const float*)d_in[4];   // [2,256] -> [512]
    const float* Wih   = (const float*)d_in[5];   // [7,2,256,128]
    const float* Whh   = (const float*)d_in[6];   // [7,2,256,64]
    const float* bvec  = (const float*)d_in[7];   // [7,2,256]
    const int*   wid   = (const int*)d_in[8];     // [64,512]

    char* ws = (char*)d_ws;
    size_t o = 0;
    unsigned short* merged = (unsigned short*)(ws + o); o += (size_t)BB * SS * DIN * 2;  // 48 MB
    float*          pre    = (float*)(ws + o);          o += (size_t)BB * SS * NG  * 4;  // 64 MB
    float*          x      = (float*)(ws + o);          o += (size_t)BB * SS * 128 * 4;  // 16 MB
    unsigned short* xbf    = (unsigned short*)(ws + o); o += (size_t)BB * SS * 128 * 2;  //  8 MB
    unsigned short* wih1b  = (unsigned short*)(ws + o); o += (size_t)NG * DIN * 2;
    unsigned short* wihb   = (unsigned short*)(ws + o); o += (size_t)7 * NG * 128 * 2;
    unsigned short* whhb   = (unsigned short*)(ws + o); o += (size_t)8 * NG * HH * 2;

    float* out  = (float*)d_out;
    float* mask = out + (size_t)BB * SS * 129;

    size_t nM = (size_t)BB * SS * DIN;
    fill_u16<<<(unsigned)((nM + 255) / 256), 256, 0, stream>>>(merged, 0, nM);
    fill_f32<<<(BB * SS + 255) / 256, 256, 0, stream>>>(mask, 0.0f, (size_t)BB * SS);
    pool_kernel<<<BB, 768, 0, stream>>>(emb, wid, merged, mask);

    auto cvt = [&](const float* src, unsigned short* dst, size_t n) {
        cvt_bf16<<<(unsigned)((n + 255) / 256), 256, 0, stream>>>(src, dst, n);
    };
    cvt(Wih1, wih1b, (size_t)NG * DIN);
    cvt(Wih,  wihb,  (size_t)7 * NG * 128);
    cvt(Whh1, whhb,  (size_t)NG * HH);
    cvt(Whh,  whhb + (size_t)NG * HH, (size_t)7 * NG * HH);

    dim3 gemmGrid(BB * SS / 128, NG / 64);     // (256, 8)
    dim3 recGrid(BB / 16, 2);                  // (4, 2)

    // layer 1: 768 -> 128
    gemm_pre<<<gemmGrid, 256, 0, stream>>>(merged, DIN, wih1b, b1, pre, DIN);
    lstm_rec<<<recGrid, 512, 0, stream>>>(pre, whhb, x, xbf, 0);

    // layers 2..8: 128 -> 128, residual from layer 3 on
    for (int l = 0; l < 7; ++l) {
        gemm_pre<<<gemmGrid, 256, 0, stream>>>(
            xbf, 128, wihb + (size_t)l * NG * 128, bvec + (size_t)l * NG, pre, 128);
        lstm_rec<<<recGrid, 512, 0, stream>>>(
            pre, whhb + (size_t)(l + 1) * NG * HH, x, xbf, (l >= 1) ? 1 : 0);
    }

    finalize_k<<<BB * SS, 128, 0, stream>>>(x, wlen, out);
}
